// MSMHA_81106162418024
// MI455X (gfx1250) — matrix operand order repaired and checked by reading, not compiled, not run on hardware
//
#include <hip/hip_runtime.h>
#include <hip/hip_bf16.h>

// Problem constants (from reference)
#define BB  2
#define SS  2048
#define DDM 512
#define HH  8
#define HD  64
#define NSC 3

typedef __attribute__((ext_vector_type(16))) __bf16 v16bf;
typedef __attribute__((ext_vector_type(8)))  __bf16 v8bf;
typedef __attribute__((ext_vector_type(8)))  float  v8f;
typedef __attribute__((ext_vector_type(4)))  unsigned int u32x4;
typedef __attribute__((ext_vector_type(8)))  int  i32x8;
typedef __attribute__((ext_vector_type(4)))  int  i32x4;

static __device__ __forceinline__ v16bf mk16(v8bf lo, v8bf hi) {
  v16bf r;
#pragma unroll
  for (int i = 0; i < 8; ++i) { r[i] = lo[i]; r[i + 8] = hi[i]; }
  return r;
}

static __device__ __forceinline__ v8bf ld8(const __bf16* p) {
  return *(const v8bf*)p;
}

static __device__ __forceinline__ v8f wmma_bf16(v16bf a, v16bf b, v8f c) {
  // (neg_a, A, neg_b, B, c_mod, C, reuse_a, reuse_b)
  return __builtin_amdgcn_wmma_f32_16x16x32_bf16(false, a, false, b,
                                                 (short)0, c, false, false);
}

// ---------------------------------------------------------------------------
// Kernel 1: xpe = bf16(x + positional_encoding)
// ---------------------------------------------------------------------------
__global__ void __launch_bounds__(256)
pe_cast_kernel(const float* __restrict__ x, __bf16* __restrict__ xpe) {
  int idx = blockIdx.x * 256 + threadIdx.x;           // covers B*S*D exactly
  int d = idx % DDM;
  int s = (idx / DDM) % SS;
  // 10000^(-c/256) = exp2(-c/256 * log2(10000))
  float ang = (float)s * exp2f(-((float)(d & 255)) * (13.287712379549449f / 256.0f));
  float pe = (d < 256) ? __sinf(ang) : __cosf(ang);
  xpe[idx] = (__bf16)(x[idx] + pe);
}

// ---------------------------------------------------------------------------
// Kernel 2: WT[m][n][k] = bf16(W_m[k][n])   (transpose so B-frags are K-contig)
// ---------------------------------------------------------------------------
__global__ void __launch_bounds__(256)
wcast_kernel(const float* __restrict__ Wq, const float* __restrict__ Wk,
             const float* __restrict__ Wv, __bf16* __restrict__ WT) {
  int idx = blockIdx.x * 256 + threadIdx.x;           // covers 3*D*D
  int m = idx / (DDM * DDM);
  int r = idx % (DDM * DDM);
  int k = r / DDM;
  int n = r % DDM;
  const float* W = (m == 0) ? Wq : (m == 1) ? Wk : Wv;
  WT[(size_t)m * DDM * DDM + (size_t)n * DDM + k] = (__bf16)W[(size_t)k * DDM + n];
}

// ---------------------------------------------------------------------------
// Kernel 3: out = bf16(A @ W + bias), A: [B*S, D] bf16, WT: [N][K] bf16
// Block = 4 waves, block tile 64x64, wave tile 16x64 (4 C frags), K step 32.
// ---------------------------------------------------------------------------
__global__ void __launch_bounds__(128)
qkv_gemm_kernel(const __bf16* __restrict__ A, const __bf16* __restrict__ WT,
                const float* __restrict__ bias, __bf16* __restrict__ out) {
  int lane = threadIdx.x & 31;
  int wv   = threadIdx.x >> 5;
  int l15  = lane & 15;
  int lh   = lane >> 4;
  int mbase = blockIdx.y * 64 + wv * 16;
  int nbase = blockIdx.x * 64;

  v8f c[4] = {};
  const __bf16* Ap = A + (size_t)(mbase + l15) * DDM;

  for (int k0 = 0; k0 < DDM; k0 += 32) {
    // A frag 16x32: lane=row(l15); K = k0 + lh*16 .. k0 + lh*16 + 15 contiguous
    v16bf af = mk16(ld8(Ap + k0 + lh * 16), ld8(Ap + k0 + lh * 16 + 8));
#pragma unroll
    for (int t = 0; t < 4; ++t) {
      // B frag 32x16: lane=col; lanes0-15: K=k0..k0+15, lanes16-31: K=k0+16..k0+31
      const __bf16* Bp = WT + (size_t)(nbase + t * 16 + l15) * DDM + k0 + lh * 16;
      v16bf bfv = mk16(ld8(Bp), ld8(Bp + 8));
      c[t] = wmma_bf16(af, bfv, c[t]);
    }
  }

#pragma unroll
  for (int t = 0; t < 4; ++t) {
    int col = nbase + t * 16 + l15;
    float bv = bias[col];
#pragma unroll
    for (int r = 0; r < 8; ++r) {
      int row = mbase + lh * 8 + r;                   // C layout: VGPR r -> row
      out[(size_t)row * DDM + col] = (__bf16)(c[t][r] + bv);
    }
  }
}

// ---------------------------------------------------------------------------
// Kernel 4: flash attention for one (scale, b, h, 64-query tile).
// K tile staged by the Tensor Data Mover (TDM) with LDS padding so rows land
// at 144B stride; V tile staged transposed by the block. Online softmax.
// ---------------------------------------------------------------------------
__global__ void __launch_bounds__(128)
attn_kernel(const __bf16* __restrict__ Q, const __bf16* __restrict__ K,
            const __bf16* __restrict__ V, float* __restrict__ att) {
  __shared__ __align__(16) __bf16 sK[32 * 72];        // [key][dim], row 144B (TDM pad)
  __shared__ __align__(16) __bf16 sVT[64 * 40];       // [dim][key], row 80B
  __shared__ __align__(16) __bf16 sP[4][16 * 40];     // per-wave P tile [q][key]

  int lane = threadIdx.x & 31;
  int wv   = threadIdx.x >> 5;
  int l15  = lane & 15;
  int lh   = lane >> 4;

  int qt  = blockIdx.x;                               // 0..S/64-1
  int h   = blockIdx.y % HH;
  int b   = blockIdx.y / HH;
  int sci = blockIdx.z;                               // 0,1,2
  int sc  = 1 << sci;                                 // 1,2,4
  int L   = SS >> sci;                                // downsampled key count
  int qbase = qt * 64 + wv * 16;
  int hcol  = h * HD;

  // Preload Q A-frags for both K-steps over HD=64.
  const __bf16* Qp = Q + (size_t)(b * SS + qbase + l15) * DDM + hcol;
  v16bf qa0 = mk16(ld8(Qp + lh * 16),      ld8(Qp + 8 + lh * 16));
  v16bf qa1 = mk16(ld8(Qp + 32 + lh * 16), ld8(Qp + 40 + lh * 16));

  float mrow[8], lrow[8], alpha[8];
  v8f o[4] = {};
#pragma unroll
  for (int r = 0; r < 8; ++r) { mrow[r] = -3.0e38f; lrow[r] = 0.0f; }

  int key_c = threadIdx.x >> 2;                       // cooperative: key row 0..31
  int doff  = (threadIdx.x & 3) * 16;                 // dim chunk 0/16/32/48

  // Invariant parts of the K-tile TDM descriptor (D#).
  unsigned sk_lds = (unsigned)(size_t)&sK[0];         // LDS byte offset of sK
  unsigned long long krow_stride = (unsigned long long)(sc * DDM); // elems

  for (int j0 = 0; j0 < L; j0 += 32) {
    __syncthreads();                                  // prior tile fully consumed

    if (wv == 0) {
      // TDM: load 32x64 bf16 K tile, row stride sc*D elems, pad LDS rows
      // by 4 dwords every 32 dwords -> 144B row stride (matches sK layout).
      unsigned long long gk =
          (unsigned long long)(size_t)(K + (size_t)(b * SS + j0 * sc) * DDM + hcol);
      u32x4 g0;
      g0[0] = 1u;                                     // count=1, user descriptor
      g0[1] = sk_lds;                                 // lds_addr
      g0[2] = (unsigned)gk;                           // global_addr[31:0]
      g0[3] = (unsigned)((gk >> 32) & 0x1ffffffu) | (2u << 30); // addr[56:32]|type=2
      i32x8 g1;
      g1[0] = (int)((1u << 16)   // data_size = 2 bytes
                  | (1u << 20)   // pad_enable
                  | (4u << 22)   // pad_interval: 32 dwords
                  | (3u << 25)); // pad_amount: 4 dwords
      g1[1] = (int)(64u << 16);                       // tensor_dim0 = 64 (lo16)
      g1[2] = (int)(32u << 16);                       // dim0 hi=0 | tensor_dim1 = 32
      g1[3] = (int)(64u << 16);                       // dim1 hi=0 | tile_dim0 = 64
      g1[4] = (int)32u;                               // tile_dim1 = 32 | tile_dim2 = 0
      g1[5] = (int)(unsigned)krow_stride;             // tensor_dim0_stride lo32
      g1[6] = (int)(unsigned)(krow_stride >> 32);     // stride0 hi16 | stride1 lo16
      g1[7] = 0;
      i32x4 z4 = {0, 0, 0, 0};
      i32x8 z8 = {0, 0, 0, 0, 0, 0, 0, 0};
      __builtin_amdgcn_tensor_load_to_lds(g0, g1, z4, z4, z8, 0);
      __builtin_amdgcn_s_wait_tensorcnt(0);           // tile complete before barrier
    }

    {
      int skey = (j0 + key_c) * sc;                   // strided downsample
      const __bf16* vp = V + (size_t)(b * SS + skey) * DDM + hcol + doff;
      v8bf v0 = ld8(vp), v1 = ld8(vp + 8);
#pragma unroll
      for (int e = 0; e < 8; ++e) {                   // transpose V into LDS
        sVT[(doff + e) * 40 + key_c]     = v0[e];
        sVT[(doff + 8 + e) * 40 + key_c] = v1[e];
      }
      if (j0 + 32 < L) {                              // prefetch next V tile row
        const __bf16* vnext = V + (size_t)(b * SS + (j0 + 32 + key_c) * sc) * DDM + hcol + doff;
        __builtin_prefetch((const void*)vnext, 0, 1);
      }
    }
    __syncthreads();

    // Scores: two 16x16 frags (keys j0..+15 and j0+16..+31), K-dim = HD = 2x32.
    v8f s0 = {}, s1 = {};
    {
      const __bf16* kb0 = &sK[(size_t)l15 * 72 + lh * 16];
      s0 = wmma_bf16(qa0, mk16(ld8(kb0),      ld8(kb0 + 8)),  s0);
      s0 = wmma_bf16(qa1, mk16(ld8(kb0 + 32), ld8(kb0 + 40)), s0);
      const __bf16* kb1 = &sK[(size_t)(16 + l15) * 72 + lh * 16];
      s1 = wmma_bf16(qa0, mk16(ld8(kb1),      ld8(kb1 + 8)),  s1);
      s1 = wmma_bf16(qa1, mk16(ld8(kb1 + 32), ld8(kb1 + 40)), s1);
    }

    // Online softmax. C-layout: VGPR r holds row m = lh*8+r, col = l15, so each
    // q-row's 16 entries live in one 16-lane half -> xor masks 1,2,4,8.
#pragma unroll
    for (int r = 0; r < 8; ++r) {
      float a0 = s0[r] * 0.125f;                      // HD^-0.5
      float a1 = s1[r] * 0.125f;
      float tmax = fmaxf(a0, a1);
#pragma unroll
      for (int msk = 1; msk < 16; msk <<= 1)
        tmax = fmaxf(tmax, __shfl_xor(tmax, msk, 32));
      float mnew = fmaxf(mrow[r], tmax);
      float al = __expf(mrow[r] - mnew);
      float p0 = __expf(a0 - mnew);
      float p1 = __expf(a1 - mnew);
      float psum = p0 + p1;
#pragma unroll
      for (int msk = 1; msk < 16; msk <<= 1)
        psum += __shfl_xor(psum, msk, 32);
      lrow[r] = lrow[r] * al + psum;
      mrow[r] = mnew;
      alpha[r] = al;
      int m = lh * 8 + r;
      sP[wv][m * 40 + l15]      = (__bf16)p0;         // C-layout -> LDS
      sP[wv][m * 40 + 16 + l15] = (__bf16)p1;
    }
#pragma unroll
    for (int t = 0; t < 4; ++t)
#pragma unroll
      for (int r = 0; r < 8; ++r) o[t][r] *= alpha[r];

    // Ensure P stores complete before re-reading in A-layout (same wave, DS in order).
    asm volatile("s_wait_dscnt 0" ::: "memory");

    // P A-frag 16x32: lane=row(l15); keys lh*16..lh*16+15 contiguous
    const __bf16* pp = &sP[wv][l15 * 40 + lh * 16];
    v16bf pa = mk16(*(const v8bf*)pp, *(const v8bf*)(pp + 8));

#pragma unroll
    for (int t = 0; t < 4; ++t) {
      // V B-frag 32x16: lane=dim col; keys contiguous from transposed LDS tile.
      const __bf16* vb = &sVT[(size_t)(t * 16 + l15) * 40 + lh * 16];
      v16bf bfv = mk16(ld8(vb), ld8(vb + 8));
      o[t] = wmma_bf16(pa, bfv, o[t]);
    }
  }

  float* ap = att + (size_t)sci * BB * SS * DDM;
#pragma unroll
  for (int t = 0; t < 4; ++t) {
#pragma unroll
    for (int r = 0; r < 8; ++r) {
      int row = qbase + lh * 8 + r;
      ap[(size_t)(b * SS + row) * DDM + hcol + t * 16 + l15] = o[t][r] / lrow[r];
    }
  }
}

// ---------------------------------------------------------------------------
// Kernel 5: mean over the 3 scales.
// ---------------------------------------------------------------------------
__global__ void __launch_bounds__(256)
mean3_kernel(const float* __restrict__ att, float* __restrict__ out) {
  const size_t N = (size_t)BB * SS * DDM;
  size_t i = (size_t)blockIdx.x * 256 + threadIdx.x;
  out[i] = (att[i] + att[i + N] + att[i + 2 * N]) * (1.0f / 3.0f);
}

// ---------------------------------------------------------------------------
extern "C" void kernel_launch(void* const* d_in, const int* in_sizes, int n_in,
                              void* d_out, int out_size, void* d_ws, size_t ws_size,
                              hipStream_t stream) {
  const float* x  = (const float*)d_in[0];
  const float* Wq = (const float*)d_in[1];
  const float* bq = (const float*)d_in[2];
  const float* Wk = (const float*)d_in[3];
  const float* bk = (const float*)d_in[4];
  const float* Wv = (const float*)d_in[5];
  const float* bv = (const float*)d_in[6];
  float* out = (float*)d_out;

  const size_t N = (size_t)BB * SS * DDM;             // 2,097,152 elems
  char* ws = (char*)d_ws;
  size_t off = 0;
  auto take = [&](size_t bytes) { char* p = ws + off; off = (off + bytes + 255) & ~(size_t)255; return p; };

  __bf16* xpe = (__bf16*)take(N * sizeof(__bf16));
  __bf16* WT  = (__bf16*)take((size_t)3 * DDM * DDM * sizeof(__bf16));
  __bf16* Qb  = (__bf16*)take(N * sizeof(__bf16));
  __bf16* Kb  = (__bf16*)take(N * sizeof(__bf16));
  __bf16* Vb  = (__bf16*)take(N * sizeof(__bf16));
  float*  att = (float*)take((size_t)NSC * N * sizeof(float));
  (void)ws_size; (void)in_sizes; (void)n_in; (void)out_size;

  pe_cast_kernel<<<(unsigned)(N / 256), 256, 0, stream>>>(x, xpe);
  wcast_kernel<<<(3 * DDM * DDM) / 256, 256, 0, stream>>>(Wq, Wk, Wv, WT);

  dim3 gg(DDM / 64, (BB * SS) / 64);
  qkv_gemm_kernel<<<gg, 128, 0, stream>>>(xpe, WT,                         bq, Qb);
  qkv_gemm_kernel<<<gg, 128, 0, stream>>>(xpe, WT + (size_t)DDM * DDM,     bk, Kb);
  qkv_gemm_kernel<<<gg, 128, 0, stream>>>(xpe, WT + (size_t)2 * DDM * DDM, bv, Vb);

  dim3 ga(SS / 64, BB * HH, NSC);
  attn_kernel<<<ga, 128, 0, stream>>>(Qb, Kb, Vb, att);

  mean3_kernel<<<(unsigned)(N / 256), 256, 0, stream>>>(att, out);
}
